// GatedDeltaNet_39496519254628
// MI455X (gfx1250) — compile-verified, matrix operand order checked by reading, not run on hardware
//
#include <hip/hip_runtime.h>
#include <math.h>

// ---------------- problem constants ----------------
constexpr int Bc    = 2;
constexpr int Sc    = 2048;
constexpr int Dc    = 2048;
constexpr int Hc    = 16;
constexpr int DKc   = 128;
constexpr int DVc   = 128;
constexpr int KDIM  = Hc * DKc;          // 2048
constexpr int VDIM  = Hc * DVc;          // 2048
constexpr int CONVC = 2 * KDIM + VDIM;   // 6144
constexpr int BS    = Bc * Sc;           // 4096

// ---------------- WMMA types ----------------
typedef __attribute__((ext_vector_type(16))) __bf16 v16bf;
typedef __attribute__((ext_vector_type(8)))  __bf16 v8bf;
typedef __attribute__((ext_vector_type(8)))  float  v8f;

union FragAB { v16bf v; v8bf h[2]; };

// Pack two fp32 -> bf16x2 (round-half-up + byte-splice via v_perm_b32).
__device__ __forceinline__ unsigned int pack_bf16(float lo, float hi) {
  unsigned int ul = __builtin_bit_cast(unsigned int, lo) + 0x8000u;
  unsigned int uh = __builtin_bit_cast(unsigned int, hi) + 0x8000u;
  // result bytes {uh.b3, uh.b2, ul.b3, ul.b2}
  return __builtin_amdgcn_perm(uh, ul, 0x07060302u);
}

// =====================================================================
// Generic TN GEMM: C[M,N] = (A .* rowscale) @ W^T
// A: [M,K] row-major fp32, W: [N,K] row-major fp32, C: [M,N] fp32.
// Block tile 128x128, BK=32, 256 threads = 8 waves (4 M-slabs x 2 N-slabs),
// each wave does a 32x64 patch via 2x4 v_wmma_f32_16x16x32_bf16 fragments.
// Staging: float4 global loads -> packed bf16 -> 8-byte LDS stores.
// =====================================================================
#define BM 128
#define BN 128
#define BK 32

__global__ __launch_bounds__(256)
void gemm_bf16_tn(const float* __restrict__ A, const float* __restrict__ W,
                  const float* __restrict__ rowscale, float* __restrict__ C,
                  int M, int N, int K) {
  __shared__ __align__(16) unsigned short sA[BM * BK];
  __shared__ __align__(16) unsigned short sW[BN * BK];

  const int tid   = threadIdx.x;
  const int lane  = tid & 31;
  const int wid   = tid >> 5;     // 0..7
  const int waveM = wid >> 1;     // 0..3 : 32-row slab
  const int waveN = wid & 1;      // 0..1 : 64-col slab
  const int blockM = blockIdx.y * BM;
  const int blockN = blockIdx.x * BN;
  const int lrow = lane & 15;     // row/col within 16x16 tile
  const int lhi  = lane >> 4;     // half-select for K packing / M rows of C

  // Fixed staging coordinates per thread (float4 granularity: 1024 f4/tile).
  int srow[4], scol[4];
  float rs[4];
  #pragma unroll
  for (int i = 0; i < 4; ++i) {
    int e = tid + i * 256;        // 0..1023
    srow[i] = e >> 3;             // 0..127
    scol[i] = (e & 7) * 4;        // 0,4,...,28
    rs[i] = rowscale ? rowscale[blockM + srow[i]] : 1.0f;
  }

  v8f acc[2][4];
  #pragma unroll
  for (int mi = 0; mi < 2; ++mi)
    #pragma unroll
    for (int ni = 0; ni < 4; ++ni)
      #pragma unroll
      for (int e = 0; e < 8; ++e) acc[mi][ni][e] = 0.0f;

  for (int kt = 0; kt < K; kt += BK) {
    // ---- stage fp32 -> packed bf16 tiles into LDS ----
    #pragma unroll
    for (int i = 0; i < 4; ++i) {
      const float4 a4 = *reinterpret_cast<const float4*>(
          &A[(size_t)(blockM + srow[i]) * K + kt + scol[i]]);
      uint2 pa;
      pa.x = pack_bf16(a4.x * rs[i], a4.y * rs[i]);
      pa.y = pack_bf16(a4.z * rs[i], a4.w * rs[i]);
      *reinterpret_cast<uint2*>(&sA[srow[i] * BK + scol[i]]) = pa;

      const float4 w4 = *reinterpret_cast<const float4*>(
          &W[(size_t)(blockN + srow[i]) * K + kt + scol[i]]);
      uint2 pw;
      pw.x = pack_bf16(w4.x, w4.y);
      pw.y = pack_bf16(w4.z, w4.w);
      *reinterpret_cast<uint2*>(&sW[srow[i] * BK + scol[i]]) = pw;
    }
    // prefetch next K tile while WMMAs run (global_prefetch_b8)
    if (kt + BK < K) {
      #pragma unroll
      for (int i = 0; i < 4; ++i) {
        __builtin_prefetch(&A[(size_t)(blockM + srow[i]) * K + kt + BK + scol[i]], 0, 1);
        __builtin_prefetch(&W[(size_t)(blockN + srow[i]) * K + kt + BK + scol[i]], 0, 1);
      }
    }
    __syncthreads();

    // ---- fragments per 16-bit A layout:
    // lane<16: K 0-7 in h[0], K 16-23 in h[1]; lane>=16: K 8-15 / 24-31.
    FragAB afr[2], bfr[4];
    #pragma unroll
    for (int mi = 0; mi < 2; ++mi) {
      const unsigned short* p = &sA[(waveM * 32 + mi * 16 + lrow) * BK];
      afr[mi].h[0] = *reinterpret_cast<const v8bf*>(p + lhi * 8);
      afr[mi].h[1] = *reinterpret_cast<const v8bf*>(p + 16 + lhi * 8);
    }
    #pragma unroll
    for (int ni = 0; ni < 4; ++ni) {
      const unsigned short* p = &sW[(waveN * 64 + ni * 16 + lrow) * BK];
      bfr[ni].h[0] = *reinterpret_cast<const v8bf*>(p + lhi * 8);
      bfr[ni].h[1] = *reinterpret_cast<const v8bf*>(p + 16 + lhi * 8);
    }

    #pragma unroll
    for (int mi = 0; mi < 2; ++mi)
      #pragma unroll
      for (int ni = 0; ni < 4; ++ni)
        acc[mi][ni] = __builtin_amdgcn_wmma_f32_16x16x32_bf16(
            false, afr[mi].v, false, bfr[ni].v,
            (short)0, acc[mi][ni], false, false);
    __syncthreads();
  }

  // C layout: col = lane&15, VGPR r -> row = r + 8*(lane>>4)
  #pragma unroll
  for (int mi = 0; mi < 2; ++mi) {
    int gm = blockM + waveM * 32 + mi * 16 + lhi * 8;
    #pragma unroll
    for (int ni = 0; ni < 4; ++ni) {
      int gn = blockN + waveN * 64 + ni * 16 + lrow;
      #pragma unroll
      for (int r = 0; r < 8; ++r)
        C[(size_t)(gm + r) * N + gn] = acc[mi][ni][r];
    }
  }
}

// =====================================================================
// beta / g projections: per (b,s) row, 32 dot products of length D
// =====================================================================
__global__ __launch_bounds__(256)
void bg_proj(const float* __restrict__ hs, const float* __restrict__ mask,
             const float* __restrict__ Wb, const float* __restrict__ Wa,
             const float* __restrict__ dt_bias, const float* __restrict__ A_log,
             float* __restrict__ betaO, float* __restrict__ gO) {
  __shared__ float xs[Dc];
  __shared__ float red[32][8];
  const int row = blockIdx.x;           // 0..BS-1
  const int tid = threadIdx.x;
  const float mv = mask[row];
  #pragma unroll
  for (int i = 0; i < Dc / 256; ++i)
    xs[tid + i * 256] = hs[(size_t)row * Dc + tid + i * 256] * mv;
  __syncthreads();

  const int oidx = tid >> 3;            // 0..31
  const int sub  = tid & 7;
  const float* Wrow = (oidx < 16) ? (Wb + (size_t)oidx * Dc)
                                  : (Wa + (size_t)(oidx - 16) * Dc);
  float p = 0.f;
  for (int d = sub; d < Dc; d += 8) p += xs[d] * Wrow[d];
  red[oidx][sub] = p;
  __syncthreads();
  if (sub == 0) {
    float dot = 0.f;
    #pragma unroll
    for (int j = 0; j < 8; ++j) dot += red[oidx][j];
    if (oidx < 16) {
      betaO[(size_t)row * Hc + oidx] = 1.f / (1.f + __expf(-dot));
    } else {
      int hh = oidx - 16;
      float xr = dot + dt_bias[hh];
      float sp = (xr > 20.f) ? xr : log1pf(__expf(xr));
      gO[(size_t)row * Hc + hh] = -__expf(A_log[hh]) * sp;
    }
  }
}

// =====================================================================
// Causal depthwise conv(K=4) + bias + SiLU + per-head L2 norm of q,k
// One (b,s) row per block.  qn scaled by DK^-0.5.
// =====================================================================
__global__ __launch_bounds__(256)
void conv_silu_norm(const float* __restrict__ mixedRaw,
                    const float* __restrict__ conv_w,
                    const float* __restrict__ conv_b,
                    float* __restrict__ qn, float* __restrict__ kn,
                    float* __restrict__ vout) {
  __shared__ float smem[CONVC];
  __shared__ float norms[32];
  const int tid = threadIdx.x;
  const int b = blockIdx.x / Sc;
  const int s = blockIdx.x % Sc;

  #pragma unroll
  for (int it = 0; it < CONVC / 256; ++it) {
    int c = it * 256 + tid;
    const float4 w4 = *reinterpret_cast<const float4*>(&conv_w[c * 4]);
    float acc = conv_b[c];
    const float wj[4] = {w4.x, w4.y, w4.z, w4.w};
    #pragma unroll
    for (int j = 0; j < 4; ++j) {
      int t = s - 3 + j;
      if (t >= 0)
        acc += wj[j] * mixedRaw[((size_t)(b * Sc + t)) * CONVC + c];
    }
    float sg = 1.f / (1.f + __expf(-acc));
    smem[c] = acc * sg;                 // SiLU
  }
  __syncthreads();

  if (tid < 32) {
    int base = (tid < 16) ? tid * DKc : (KDIM + (tid - 16) * DKc);
    float ss = 0.f;
    for (int i = 0; i < DKc; ++i) { float v = smem[base + i]; ss += v * v; }
    norms[tid] = rsqrtf(ss + 1e-6f);
  }
  __syncthreads();

  const size_t rb = (size_t)(b * Sc + s) * KDIM;  // KDIM == VDIM == 2048
  const float qscale = 0.08838834764831845f;      // DK^-0.5
  #pragma unroll
  for (int it = 0; it < CONVC / 256; ++it) {
    int c = it * 256 + tid;
    float v = smem[c];
    if (c < KDIM) {
      qn[rb + c] = v * norms[c >> 7] * qscale;
    } else if (c < 2 * KDIM) {
      int cc = c - KDIM;
      kn[rb + cc] = v * norms[16 + (cc >> 7)];
    } else {
      vout[rb + (c - 2 * KDIM)] = v;
    }
  }
}

// =====================================================================
// Gated delta-rule scan.  One (b,h) per block, 256 threads:
// col = tid&127 owns state column v=col; half = tid>>7 owns 64 of 128 k's.
// State kept entirely in VGPRs (64 fp32/lane).  3 barriers / step.
// =====================================================================
__global__ __launch_bounds__(256)
void delta_scan(const float* __restrict__ qn, const float* __restrict__ kn,
                const float* __restrict__ vv, const float* __restrict__ gB,
                const float* __restrict__ betaB, float* __restrict__ ob) {
  __shared__ float q_sh[DKc];
  __shared__ float k_sh[DKc];
  __shared__ float redKV[256];
  __shared__ float redO[256];
  const int tid  = threadIdx.x;
  const int col  = tid & 127;
  const int half = tid >> 7;            // 0/1
  const int b = blockIdx.x >> 4;
  const int h = blockIdx.x & 15;

  float st[64];
  #pragma unroll
  for (int j = 0; j < 64; ++j) st[j] = 0.f;

  for (int s = 0; s < Sc; ++s) {
    const size_t rbase = (size_t)(b * Sc + s);
    const size_t vec   = rbase * KDIM + h * DKc;
    if (tid < 128) q_sh[tid] = qn[vec + tid];
    else           k_sh[tid - 128] = kn[vec + (tid - 128)];
    const float gv = gB[rbase * Hc + h];
    const float bv = betaB[rbase * Hc + h];
    const float vval = vv[vec + col];
    __syncthreads();                                   // A

    const float decay = __expf(gv);
    float kvp = 0.f;
    #pragma unroll
    for (int j = 0; j < 64; ++j) kvp += k_sh[half * 64 + j] * st[j];
    redKV[tid] = kvp;
    __syncthreads();                                   // B
    const float kv = (redKV[col] + redKV[col + 128]) * decay;
    const float delta = (vval - kv) * bv;
    float op = 0.f;
    #pragma unroll
    for (int j = 0; j < 64; ++j) {
      st[j] = st[j] * decay + k_sh[half * 64 + j] * delta;
      op += q_sh[half * 64 + j] * st[j];
    }
    redO[tid] = op;
    __syncthreads();                                   // C
    if (tid < 128) ob[vec + tid] = redO[tid] + redO[tid + 128];
    // barrier C guards q_sh/k_sh/redKV reuse by the next iteration;
    // redO rewrite is separated by next A and B barriers.
  }
}

// =====================================================================
// Gated RMSNorm: o * rsqrt(mean(o^2)+eps) * (1+w) * silu(z)
// =====================================================================
__global__ __launch_bounds__(256)
void norm_gate(const float* __restrict__ ob, const float* __restrict__ zb,
               const float* __restrict__ nw, float* __restrict__ og) {
  __shared__ float red[256];
  __shared__ float inv[Hc];
  const int row = blockIdx.x;
  const int tid = threadIdx.x;
  const int hh = tid >> 4;              // head, 16 threads per head
  const int i  = tid & 15;
  const size_t base = (size_t)row * VDIM + hh * DVc;

  float vals[8];
  float ss = 0.f;
  #pragma unroll
  for (int j = 0; j < 8; ++j) {
    float v = ob[base + i * 8 + j];
    vals[j] = v; ss += v * v;
  }
  red[tid] = ss;
  __syncthreads();
  if (i == 0) {
    float t = 0.f;
    #pragma unroll
    for (int j = 0; j < 16; ++j) t += red[hh * 16 + j];
    inv[hh] = rsqrtf(t * (1.0f / DVc) + 1e-6f);
  }
  __syncthreads();
  const float iv = inv[hh];
  #pragma unroll
  for (int j = 0; j < 8; ++j) {
    int dv = i * 8 + j;
    float z = zb[base + dv];
    float sz = z / (1.f + __expf(-z));
    og[base + dv] = vals[j] * iv * (1.f + nw[dv]) * sz;
  }
}

// =====================================================================
extern "C" void kernel_launch(void* const* d_in, const int* in_sizes, int n_in,
                              void* d_out, int out_size, void* d_ws, size_t ws_size,
                              hipStream_t stream) {
  (void)in_sizes; (void)n_in; (void)out_size; (void)ws_size;
  const float* hs      = (const float*)d_in[0];
  const float* mask    = (const float*)d_in[1];
  const float* W_qkv   = (const float*)d_in[2];
  const float* conv_w  = (const float*)d_in[3];
  const float* conv_b  = (const float*)d_in[4];
  const float* W_z     = (const float*)d_in[5];
  const float* W_b     = (const float*)d_in[6];
  const float* W_a     = (const float*)d_in[7];
  const float* dt_bias = (const float*)d_in[8];
  const float* A_log   = (const float*)d_in[9];
  const float* norm_w  = (const float*)d_in[10];
  const float* W_out   = (const float*)d_in[11];
  float* out = (float*)d_out;

  float* ws = (float*)d_ws;
  size_t off = 0;
  float* mixedRaw = ws + off; off += (size_t)BS * CONVC;  // 25.2M floats
  float* qn       = ws + off; off += (size_t)BS * KDIM;
  float* kn       = ws + off; off += (size_t)BS * KDIM;
  float* vb       = ws + off; off += (size_t)BS * VDIM;
  float* zb       = ws + off; off += (size_t)BS * VDIM;
  float* ob       = ws + off; off += (size_t)BS * VDIM;
  float* og       = ws + off; off += (size_t)BS * VDIM;
  float* betaB    = ws + off; off += (size_t)BS * Hc;
  float* gB       = ws + off; off += (size_t)BS * Hc;

  // 1) mixed = (x*mask) @ W_qkv^T      [4096 x 6144]
  gemm_bf16_tn<<<dim3(CONVC / BN, BS / BM), 256, 0, stream>>>(
      hs, W_qkv, mask, mixedRaw, BS, CONVC, Dc);
  // 2) z = (x*mask) @ W_z^T            [4096 x 2048]
  gemm_bf16_tn<<<dim3(VDIM / BN, BS / BM), 256, 0, stream>>>(
      hs, W_z, mask, zb, BS, VDIM, Dc);
  // 3) beta, g projections
  bg_proj<<<BS, 256, 0, stream>>>(hs, mask, W_b, W_a, dt_bias, A_log, betaB, gB);
  // 4) conv + silu + l2norm split
  conv_silu_norm<<<BS, 256, 0, stream>>>(mixedRaw, conv_w, conv_b, qn, kn, vb);
  // 5) gated delta-rule scan
  delta_scan<<<Bc * Hc, 256, 0, stream>>>(qn, kn, vb, gB, betaB, ob);
  // 6) RMSNorm + gate
  norm_gate<<<BS, 256, 0, stream>>>(ob, zb, norm_w, og);
  // 7) out = og @ W_out^T              [4096 x 2048]
  gemm_bf16_tn<<<dim3(Dc / BN, BS / BM), 256, 0, stream>>>(
      og, W_out, nullptr, out, BS, Dc, VDIM);
}
